// HieRecUserEncoder_42949673191
// MI455X (gfx1250) — compile-verified, matrix operand order checked by reading, not run on hardware
//
#include <hip/hip_runtime.h>
#include <hip/hip_bf16.h>

// ---------------- problem constants (from reference) ----------------
static constexpr int kB  = 2048;
static constexpr int kH  = 50;
static constexpr int kD  = 400;
static constexpr int kS  = 285;
static constexpr int kC  = 18;
static constexpr int kNC = 50;
static constexpr float kFILL = 1e-30f;
static constexpr int kSTILES = (kS + 15) / 16;   // 18
static constexpr int kNTILES = kD / 16;          // 25
static constexpr int kHPAD   = 72;               // stage1 LDS stride (bf16) per d-row: 144B, 16B-aligned, bank-cycling
static constexpr int kSPAD   = 40;               // stage2 LDS stride (bf16) per d-row: 80B

typedef __attribute__((ext_vector_type(16))) __bf16 v16bf;
typedef __attribute__((ext_vector_type(8)))  __bf16 v8bf;
typedef __attribute__((ext_vector_type(8)))  float  v8f;
typedef __attribute__((ext_vector_type(4)))  float  v4f;

// CDNA5 16-bit A/B matrix lane->K mapping (16x32), per ISA 7.12.2:
// lanes 0-15 hold K = {0..7, 16..23}; lanes 16-31 hold K = {8..15, 24..31}
__device__ __forceinline__ int wmma_k16(int hiK, int i) {
    return (i < 8) ? (hiK + i) : (16 + hiK + (i - 8));
}

__device__ __forceinline__ v16bf cat8(v8bf lo, v8bf hi) {
    return __builtin_shufflevector(lo, hi, 0, 1, 2, 3, 4, 5, 6, 7,
                                           8, 9, 10, 11, 12, 13, 14, 15);
}

// =====================================================================
// Kernel 1: per-batch prep. news scores, per-s softmax stats, counts,
// per-s category bitmask, sub_weights output.
// =====================================================================
__global__ void k_prep(const float* __restrict__ vectors,
                       const int*   __restrict__ subcat,
                       const int*   __restrict__ category,
                       const float* __restrict__ w_news,
                       float* __restrict__ ws_ns,      // [B,H]
                       float* __restrict__ ws_m,       // [B,S]
                       float* __restrict__ ws_dn,      // [B,S]
                       int*   __restrict__ ws_scnt,    // [B,S]
                       unsigned* __restrict__ ws_cmask,// [B,S]
                       int*   __restrict__ ws_ccnt,    // [B,C]
                       float* __restrict__ o_subw)     // [B,S]
{
    const int b   = blockIdx.x;
    const int tid = threadIdx.x;
    __shared__ int   sh_sub[kH];
    __shared__ int   sh_cat[kH];
    __shared__ float sh_score[kH];

    if (tid < kH) {
        sh_sub[tid] = subcat[b * kH + tid];
        sh_cat[tid] = category[b * kH + tid];
    }
    __syncthreads();

    // news_scores: one wave per history item, float4 loads
    const int lane = tid & 31, wave = tid >> 5;
    const v4f* w4 = (const v4f*)w_news;
    for (int h = wave; h < kH; h += 8) {
        const v4f* v4 = (const v4f*)(vectors + ((size_t)b * kH + h) * kD);
        float sum = 0.f;
        for (int q = lane; q < kD / 4; q += 32) {
            v4f a = v4[q], w = w4[q];
            sum += a[0] * w[0] + a[1] * w[1] + a[2] * w[2] + a[3] * w[3];
        }
        #pragma unroll
        for (int off = 16; off > 0; off >>= 1) sum += __shfl_xor(sum, off, 32);
        if (lane == 0) {
            sh_score[h] = sum;
            ws_ns[b * kH + h] = sum;
        }
    }
    __syncthreads();

    // per-subcategory stats (softmax over h includes FILL entries)
    for (int s = tid; s < kS; s += blockDim.x) {
        int cnt = 0; unsigned msk = 0u;
        float mx = -3.4e38f;
        for (int h = 0; h < kH; ++h) {
            bool hit = (sh_sub[h] == s);
            if (hit) { ++cnt; msk |= (1u << sh_cat[h]); }
            float x = hit ? sh_score[h] : kFILL;
            mx = fmaxf(mx, x);
        }
        float den = 0.f;
        for (int h = 0; h < kH; ++h) {
            float x = (sh_sub[h] == s) ? sh_score[h] : kFILL;
            den += __expf(x - mx);
        }
        ws_m[(size_t)b * kS + s]     = mx;
        ws_dn[(size_t)b * kS + s]    = den;
        ws_scnt[(size_t)b * kS + s]  = cnt;
        ws_cmask[(size_t)b * kS + s] = msk;
        o_subw[(size_t)b * kS + s]   = (float)cnt * (1.0f / (float)kH);
    }
    // per-category counts
    for (int c = tid; c < kC; c += blockDim.x) {
        int cnt = 0;
        for (int h = 0; h < kH; ++h) cnt += (sh_cat[h] == c);
        ws_ccnt[b * kC + c] = cnt;
    }
}

// =====================================================================
// Kernel 2: stage-1 GEMM, one block per batch, 512 threads (16 waves).
// vectors[b] staged once into transposed bf16 LDS (float4 fill);
// subcat_embed enters through the WMMA C operand; epilogue = store + ds_add.
// =====================================================================
__global__ void __launch_bounds__(512, 1)
k_stage1(const float* __restrict__ vectors,
         const int*   __restrict__ subcat,
         const float* __restrict__ subcat_embed,
         const float* __restrict__ clicks_embed,
         const float* __restrict__ w_subcat,
         const float* __restrict__ ws_ns,
         const float* __restrict__ ws_m,
         const float* __restrict__ ws_dn,
         const int*   __restrict__ ws_scnt,
         float* __restrict__ ws_sc,      // [B,S] sc_scores out
         float* __restrict__ o_sub)      // [B,S,D]
{
    const int b   = blockIdx.x;
    const int tid = threadIdx.x;

    __shared__ __align__(16) __bf16 sh_vec[kD * kHPAD]; // [d][h] transposed, 57.6KB
    __shared__ float sh_score[64];
    __shared__ int   sh_sub[64];
    __shared__ float sh_m[288], sh_rdn[288], sh_scacc[288];
    __shared__ float sh_wsub[kD];

    if (tid < 64) {
        sh_score[tid] = (tid < kH) ? ws_ns[b * kH + tid] : 0.f;
        sh_sub[tid]   = (tid < kH) ? subcat[b * kH + tid] : -1;
    }
    for (int s = tid; s < 288; s += 512) {
        bool v = (s < kS);
        sh_m[s]     = v ? ws_m[(size_t)b * kS + s] : 0.f;
        sh_rdn[s]   = v ? (1.0f / ws_dn[(size_t)b * kS + s]) : 0.f;
        sh_scacc[s] = 0.f;
    }
    if (tid < kD) sh_wsub[tid] = w_subcat[tid];

    // stage vectors[b] -> LDS, transposed [d][h], f32 -> bf16, float4 loads
    const v4f* vb4 = (const v4f*)(vectors + (size_t)b * kH * kD);
    for (int idx = tid; idx < 64 * (kD / 4); idx += 512) {
        int h = idx / (kD / 4);
        int q = idx - h * (kD / 4);
        v4f v = {};
        if (h < kH) v = vb4[(size_t)h * (kD / 4) + q];
        int d = q * 4;
        sh_vec[(d + 0) * kHPAD + h] = (__bf16)v[0];
        sh_vec[(d + 1) * kHPAD + h] = (__bf16)v[1];
        sh_vec[(d + 2) * kHPAD + h] = (__bf16)v[2];
        sh_vec[(d + 3) * kHPAD + h] = (__bf16)v[3];
    }
    __syncthreads();

    const int lane = tid & 31, wave = tid >> 5;
    const int hiK  = (lane & 16) ? 8 : 0;
    const int mrow = lane & 15;
    const int ncol = lane & 15;
    const int hiM  = (lane & 16) ? 8 : 0;

    for (int st = 0; st < kSTILES; ++st) {
        const int s0 = st * 16;
        const int s  = s0 + mrow;
        const bool sv = (s < kS);
        const bool full = (s0 + 16 <= kS);
        const float ms  = sv ? sh_m[s]   : 0.f;
        const float rdn = sv ? sh_rdn[s] : 0.f;

        // A: rebuilt attention weights, bf16, K = h padded to 64
        v16bf A0, A1;
        #pragma unroll
        for (int i = 0; i < 16; ++i) {
            int k = wmma_k16(hiK, i);
            float w0 = 0.f, w1 = 0.f;
            if (sv) {
                float x0 = (sh_sub[k] == s) ? sh_score[k] : kFILL;
                w0 = __expf(x0 - ms) * rdn;
                int h1 = k + 32;
                if (h1 < kH) {
                    float x1 = (sh_sub[h1] == s) ? sh_score[h1] : kFILL;
                    w1 = __expf(x1 - ms) * rdn;
                }
            }
            A0[i] = (__bf16)w0;
            A1[i] = (__bf16)w1;
        }

        for (int nt = wave; nt < kNTILES; nt += 16) {
            const int dd = nt * 16 + ncol;
            const float* se = subcat_embed + (size_t)s0 * kD + dd;
            float*       os = o_sub + ((size_t)b * kS + s0) * kD + dd;

            // C operand = subcat_embed tile (loads batched, overlap ds/wmma)
            v8f acc;
            if (full) {
                #pragma unroll
                for (int v = 0; v < 8; ++v) acc[v] = se[(size_t)(v + hiM) * kD];
            } else {
                #pragma unroll
                for (int v = 0; v < 8; ++v)
                    acc[v] = (s0 + v + hiM < kS) ? se[(size_t)(v + hiM) * kD] : 0.f;
            }

            const __bf16* col = &sh_vec[dd * kHPAD];
            v16bf B0 = cat8(*(const v8bf*)(col + hiK),
                            *(const v8bf*)(col + 16 + hiK));
            v16bf B1 = cat8(*(const v8bf*)(col + 32 + hiK),
                            *(const v8bf*)(col + 48 + hiK));
            acc = __builtin_amdgcn_wmma_f32_16x16x32_bf16(false, A0, false, B0,
                                                          (short)0, acc, false, false);
            acc = __builtin_amdgcn_wmma_f32_16x16x32_bf16(false, A1, false, B1,
                                                          (short)0, acc, false, false);

            const float wsub = sh_wsub[dd];
            if (full) {
                #pragma unroll
                for (int v = 0; v < 8; ++v) {
                    os[(size_t)(v + hiM) * kD] = acc[v];
                    atomicAdd(&sh_scacc[s0 + v + hiM], acc[v] * wsub); // ds_add_f32
                }
            } else {
                #pragma unroll
                for (int v = 0; v < 8; ++v) {
                    if (s0 + v + hiM < kS) {
                        os[(size_t)(v + hiM) * kD] = acc[v];
                        atomicAdd(&sh_scacc[s0 + v + hiM], acc[v] * wsub);
                    }
                }
            }
        }
    }
    __syncthreads();

    // finish sc_scores: add clicks-embedding dot with w_subcat[D:]
    for (int ss = tid; ss < kS; ss += 512) {
        int cnt = ws_scnt[(size_t)b * kS + ss];
        if (cnt > kH - 1) cnt = kH - 1;   // JAX gather clamps OOB
        float t = sh_scacc[ss];
        for (int j = 0; j < kNC; ++j)
            t += clicks_embed[cnt * kNC + j] * w_subcat[kD + j];
        ws_sc[(size_t)b * kS + ss] = t;
    }
}

// =====================================================================
// Kernel 3: stage-2 + stage-3, one block per batch, 512 threads.
// K-chunked: sub_repr chunk -> transposed bf16 LDS (float4 fill + prefetch
// of the next chunk); accumulators pre-loaded with cat_embed (C operand);
// chunk buffer reused afterwards to hold cat_repr for user_repr.
// =====================================================================
__global__ void __launch_bounds__(512, 1)
k_stage2(const float* __restrict__ o_sub,      // [B,S,D] (L2-hot)
         const float* __restrict__ cat_embed,
         const float* __restrict__ clicks_embed,
         const float* __restrict__ w_cat,
         const float* __restrict__ ws_sc,
         const unsigned* __restrict__ ws_cmask,
         const int*   __restrict__ ws_ccnt,
         float* __restrict__ o_cat,             // [B,C,D]
         float* __restrict__ o_catw,            // [B,C]
         float* __restrict__ o_user)            // [B,D]
{
    const int b   = blockIdx.x;
    const int tid = threadIdx.x;

    __shared__ __align__(16) unsigned char sh_buf[kD * kSPAD * 2]; // 32000B
    __shared__ float    sh_sc[288];
    __shared__ unsigned sh_msk[288];
    __shared__ float sh_mc[32], sh_rdn[32], sh_cs[32], sh_cw[32];
    __shared__ int   sh_cc[32];
    __shared__ float sh_wcat[kD];

    __bf16* chunk  = (__bf16*)sh_buf;     // [d][32 padded to 40] during GEMM
    float*  catbuf = (float*)sh_buf;      // [C][D] (28800B) after GEMM

    for (int s = tid; s < 288; s += 512) {
        bool v = (s < kS);
        sh_sc[s]  = v ? ws_sc[(size_t)b * kS + s] : 0.f;
        sh_msk[s] = v ? ws_cmask[(size_t)b * kS + s] : 0u;
    }
    if (tid < 32) { sh_cs[tid] = 0.f; sh_cc[tid] = 0; }
    if (tid < kC) sh_cc[tid] = ws_ccnt[b * kC + tid];
    if (tid < kD) sh_wcat[tid] = w_cat[tid];
    __syncthreads();

    // per-category softmax stats over s (masked entries -> FILL, still counted)
    if (tid < kC) {
        float mx = -3.4e38f;
        for (int s2 = 0; s2 < kS; ++s2) {
            float x = ((sh_msk[s2] >> tid) & 1u) ? sh_sc[s2] : kFILL;
            mx = fmaxf(mx, x);
        }
        float den = 0.f;
        for (int s2 = 0; s2 < kS; ++s2) {
            float x = ((sh_msk[s2] >> tid) & 1u) ? sh_sc[s2] : kFILL;
            den += __expf(x - mx);
        }
        sh_mc[tid]  = mx;
        sh_rdn[tid] = 1.0f / den;
    }

    const int lane = tid & 31, wave = tid >> 5;
    const int hiK  = (lane & 16) ? 8 : 0;
    const int mrow = lane & 15;
    const int ncol = lane & 15;
    const int hiM  = (lane & 16) ? 8 : 0;
    const float* subr = o_sub + (size_t)b * kS * kD;
    const v4f* subr4  = (const v4f*)subr;

    // accumulators pre-loaded with cat_embed (WMMA C operand), once
    v8f acc[4] = {{}, {}, {}, {}};
    #pragma unroll
    for (int j = 0; j < 4; ++j) {
        int t = wave + 16 * j;
        if (t < 2 * kNTILES) {
            int mt = t / kNTILES;
            int nt = t - mt * kNTILES;
            int dd = nt * 16 + ncol;
            #pragma unroll
            for (int v = 0; v < 8; ++v) {
                int cc = mt * 16 + v + hiM;
                acc[j][v] = (cc < kC) ? cat_embed[(size_t)cc * kD + dd] : 0.f;
            }
        }
    }

    for (int kc = 0; kc < 9; ++kc) {      // K = 285 padded to 288
        const int kbase = kc * 32;
        __syncthreads();                  // prior chunk fully consumed
        for (int idx = tid; idx < 32 * (kD / 4); idx += 512) {
            int sl = idx / (kD / 4);
            int q  = idx - sl * (kD / 4);
            int s2 = kbase + sl;
            v4f v = {};
            if (s2 < kS) v = subr4[(size_t)s2 * (kD / 4) + q];
            int d = q * 4;
            chunk[(d + 0) * kSPAD + sl] = (__bf16)v[0];
            chunk[(d + 1) * kSPAD + sl] = (__bf16)v[1];
            chunk[(d + 2) * kSPAD + sl] = (__bf16)v[2];
            chunk[(d + 3) * kSPAD + sl] = (__bf16)v[3];
            int s3 = kbase + 32 + sl;     // pull next chunk toward the caches
            if (s3 < kS)
                __builtin_prefetch(&subr4[(size_t)s3 * (kD / 4) + q], 0, 0);
        }
        __syncthreads();

        // A for both m-tiles (c = 0..15, 16..31; valid c < 18)
        v16bf Amt[2];
        #pragma unroll
        for (int mt = 0; mt < 2; ++mt) {
            int c = mt * 16 + mrow;
            bool cv = (c < kC);
            float mc = cv ? sh_mc[c]  : 0.f;
            float rd = cv ? sh_rdn[c] : 0.f;
            #pragma unroll
            for (int i = 0; i < 16; ++i) {
                int s2 = kbase + wmma_k16(hiK, i);
                float a = 0.f;
                if (cv && s2 < kS) {
                    float x = ((sh_msk[s2] >> c) & 1u) ? sh_sc[s2] : kFILL;
                    a = __expf(x - mc) * rd;
                }
                Amt[mt][i] = (__bf16)a;
            }
        }

        #pragma unroll
        for (int j = 0; j < 4; ++j) {
            int t = wave + 16 * j;
            if (t < 2 * kNTILES) {
                int mt = t / kNTILES;
                int nt = t - mt * kNTILES;
                int dd = nt * 16 + ncol;
                const __bf16* col = &chunk[dd * kSPAD];
                v16bf Bv = cat8(*(const v8bf*)(col + hiK),
                                *(const v8bf*)(col + 16 + hiK));
                acc[j] = __builtin_amdgcn_wmma_f32_16x16x32_bf16(
                    false, Amt[mt], false, Bv, (short)0, acc[j], false, false);
            }
        }
    }
    __syncthreads();                      // GEMM reads done; reuse sh_buf

    #pragma unroll
    for (int j = 0; j < 4; ++j) {
        int t = wave + 16 * j;
        if (t < 2 * kNTILES) {
            int mt = t / kNTILES;
            int nt = t - mt * kNTILES;
            int dd = nt * 16 + ncol;
            #pragma unroll
            for (int v = 0; v < 8; ++v) {
                int cc = mt * 16 + v + hiM;
                if (cc < kC) {
                    float val = acc[j][v];
                    o_cat[((size_t)b * kC + cc) * kD + dd] = val;
                    catbuf[cc * kD + dd] = val;
                    atomicAdd(&sh_cs[cc], val * sh_wcat[dd]);   // ds_add_f32
                }
            }
        }
    }
    __syncthreads();

    // c_scores += clicks term; cat_weights out
    if (tid < kC) {
        int cnt = sh_cc[tid];
        o_catw[b * kC + tid] = (float)cnt * (1.0f / (float)kH);
        if (cnt > kH - 1) cnt = kH - 1;
        float t = sh_cs[tid];
        for (int j = 0; j < kNC; ++j)
            t += clicks_embed[cnt * kNC + j] * w_cat[kD + j];
        sh_cs[tid] = t;
    }
    __syncthreads();

    // softmax over C
    if (tid < kC) {
        float mx = -3.4e38f;
        for (int c2 = 0; c2 < kC; ++c2) mx = fmaxf(mx, sh_cs[c2]);
        float den = 0.f;
        for (int c2 = 0; c2 < kC; ++c2) den += __expf(sh_cs[c2] - mx);
        sh_cw[tid] = __expf(sh_cs[tid] - mx) / den;
    }
    __syncthreads();

    // user_repr from LDS-resident cat_repr
    for (int d = tid; d < kD; d += 512) {
        float u = 0.f;
        for (int c2 = 0; c2 < kC; ++c2)
            u += sh_cw[c2] * catbuf[c2 * kD + d];
        o_user[(size_t)b * kD + d] = u;
    }
}

// =====================================================================
extern "C" void kernel_launch(void* const* d_in, const int* in_sizes, int n_in,
                              void* d_out, int out_size, void* d_ws, size_t ws_size,
                              hipStream_t stream) {
    (void)in_sizes; (void)n_in; (void)out_size; (void)ws_size;

    const float* vectors      = (const float*)d_in[0];
    const int*   subcategory  = (const int*)  d_in[1];
    const int*   category     = (const int*)  d_in[2];
    const float* subcat_embed = (const float*)d_in[3];
    const float* cat_embed    = (const float*)d_in[4];
    const float* clicks_embed = (const float*)d_in[5];
    const float* w_news       = (const float*)d_in[6];
    const float* w_subcat     = (const float*)d_in[7];
    const float* w_cat        = (const float*)d_in[8];

    // outputs, concatenated in return order
    float* out    = (float*)d_out;
    float* o_sub  = out;                                    // [B,S,D]
    float* o_subw = o_sub  + (size_t)kB * kS * kD;          // [B,S]
    float* o_cat  = o_subw + (size_t)kB * kS;               // [B,C,D]
    float* o_catw = o_cat  + (size_t)kB * kC * kD;          // [B,C]
    float* o_user = o_catw + (size_t)kB * kC;               // [B,D]

    // workspace layout (~12.2 MB)
    float*    ws_ns    = (float*)d_ws;                      // [B,H]
    float*    ws_m     = ws_ns  + (size_t)kB * kH;          // [B,S]
    float*    ws_dn    = ws_m   + (size_t)kB * kS;          // [B,S]
    float*    ws_sc    = ws_dn  + (size_t)kB * kS;          // [B,S]
    int*      ws_scnt  = (int*)(ws_sc + (size_t)kB * kS);   // [B,S]
    unsigned* ws_cmask = (unsigned*)(ws_scnt + (size_t)kB * kS); // [B,S]
    int*      ws_ccnt  = (int*)(ws_cmask + (size_t)kB * kS);     // [B,C]

    k_prep<<<kB, 256, 0, stream>>>(vectors, subcategory, category, w_news,
                                   ws_ns, ws_m, ws_dn, ws_scnt, ws_cmask,
                                   ws_ccnt, o_subw);

    k_stage1<<<kB, 512, 0, stream>>>(vectors, subcategory, subcat_embed,
                                     clicks_embed, w_subcat,
                                     ws_ns, ws_m, ws_dn, ws_scnt,
                                     ws_sc, o_sub);

    k_stage2<<<kB, 512, 0, stream>>>(o_sub, cat_embed, clicks_embed, w_cat,
                                     ws_sc, ws_cmask, ws_ccnt,
                                     o_cat, o_catw, o_user);
}
